// Attention_model_2662879723876
// MI455X (gfx1250) — matrix operand order repaired and checked by reading, not compiled, not run on hardware
//
#include <hip/hip_runtime.h>

// ---------------------------------------------------------------------------
// Bahdanau attention for MI455X (gfx1250, wave32, WMMA).
// Strategy: BF16 v_wmma_f32_16x16x32_bf16 for both GEMMs (memory-roofline
// balanced at 23.3 TB/s), f32 accumulation, fused tanh/V-dot/softmax/context.
// ---------------------------------------------------------------------------

typedef __attribute__((ext_vector_type(16))) __bf16 v16bf;
typedef __attribute__((ext_vector_type(8)))  float  v8f;

#define LDIM 64
#define DDIM 256
#define UDIM 512

// Load a 16-element bf16 fragment from two 16-byte chunks.
__device__ __forceinline__ v16bf load_frag_pair(const __bf16* p0, const __bf16* p1) {
  v16bf f;
  reinterpret_cast<uint4*>(&f)[0] = *reinterpret_cast<const uint4*>(p0);
  reinterpret_cast<uint4*>(&f)[1] = *reinterpret_cast<const uint4*>(p1);
  return f;
}

// ---------------------------------------------------------------------------
// Kernel 0: f32 -> bf16 conversion of W1, W2, hidden into workspace.
// ---------------------------------------------------------------------------
__global__ void convert_kernel(const float* __restrict__ W1,
                               const float* __restrict__ W2,
                               const float* __restrict__ hidden,
                               __bf16* __restrict__ W1bf,
                               __bf16* __restrict__ W2bf,
                               __bf16* __restrict__ hbf, int B) {
  int i = blockIdx.x * blockDim.x + threadIdx.x;
  int stride = gridDim.x * blockDim.x;
  for (int t = i; t < DDIM * UDIM; t += stride) W1bf[t] = (__bf16)W1[t];
  for (int t = i; t < UDIM * UDIM; t += stride) W2bf[t] = (__bf16)W2[t];
  const int nh = B * UDIM;
  for (int t = i; t < nh; t += stride) hbf[t] = (__bf16)hidden[t];
}

// ---------------------------------------------------------------------------
// Kernel 1: projhb[b,n] = (hidden @ W2)[b,n] + b2[n] + b1[n]
// Block = 256 threads (8 waves). Block computes 64 rows x 512 cols.
// Wave w owns columns [w*64, w*64+64): 4 M-tiles x 4 N-tiles of 16x16.
// ---------------------------------------------------------------------------
__global__ __launch_bounds__(256) void proj_hidden_kernel(
    const __bf16* __restrict__ hbf, const __bf16* __restrict__ W2bf,
    const float* __restrict__ b1, const float* __restrict__ b2,
    float* __restrict__ projhb) {
  const int tid   = threadIdx.x;
  const int wave  = tid >> 5;
  const int lane  = tid & 31;
  const int lhalf = lane & 15;
  const int koff  = (lane < 16) ? 0 : 8;     // ISA A-frag K offset for hi lanes
  const int mrow0 = blockIdx.x * 64;
  const int nbase = wave * 64;

  v8f acc[4][4] = {};
  for (int kk = 0; kk < UDIM / 32; ++kk) {
    const int k0 = kk * 32;
    v16bf afr[4];
#pragma unroll
    for (int mt = 0; mt < 4; ++mt) {
      const __bf16* rp =
          hbf + (size_t)(mrow0 + mt * 16 + lhalf) * UDIM + k0 + koff;
      afr[mt] = load_frag_pair(rp, rp + 16);          // K and K+16 chunks
    }
#pragma unroll
    for (int nt = 0; nt < 4; ++nt) {
      const __bf16* bp =
          W2bf + (size_t)(k0 + koff) * UDIM + nbase + nt * 16 + lhalf;
      v16bf bfr;
#pragma unroll
      for (int s = 0; s < 16; ++s) bfr[s] = bp[(size_t)(s + (s & 8)) * UDIM];
#pragma unroll
      for (int mt = 0; mt < 4; ++mt)
        acc[mt][nt] = __builtin_amdgcn_wmma_f32_16x16x32_bf16(
            false, afr[mt], false, bfr, (short)0, acc[mt][nt], false, false);
    }
  }

#pragma unroll
  for (int nt = 0; nt < 4; ++nt) {
    const int n = nbase + nt * 16 + lhalf;
    const float bias = b1[n] + b2[n];
#pragma unroll
    for (int mt = 0; mt < 4; ++mt)
#pragma unroll
      for (int j = 0; j < 8; ++j) {
        const int m = mrow0 + mt * 16 + j + koff;     // C layout: VGPR j -> row
        projhb[(size_t)m * UDIM + n] = acc[mt][nt][j] + bias;
      }
  }
}

// ---------------------------------------------------------------------------
// Kernel 2: per-batch fused attention. One block per batch, 256 threads.
//  - stage features[b] f32->bf16 in LDS (row stride padded to 264 bf16)
//  - 8 waves x (4 M-tiles x 4 N-tiles) WMMA over K=256
//  - epilogue: tanh(acc + projhb) dot V -> logits via shfl + ds_add_f32
//  - softmax over L=64, context = sum_l w_l * features[b,l,:]
// ---------------------------------------------------------------------------
__global__ __launch_bounds__(256) void attn_kernel(
    const float* __restrict__ features, const __bf16* __restrict__ W1bf,
    const float* __restrict__ projhb, const float* __restrict__ V,
    float* __restrict__ outCtx, float* __restrict__ outAttn) {
  __shared__ __align__(16) __bf16 fBf[LDIM][DDIM + 8];
  __shared__ float logits[LDIM];
  __shared__ float aw[LDIM];

  const int tid = threadIdx.x;
  const int b   = blockIdx.x;
  const float* fsrc = features + (size_t)b * LDIM * DDIM;

  // Stage + convert features tile.
  for (int i = tid; i < LDIM * DDIM / 4; i += 256) {
    float4 v = reinterpret_cast<const float4*>(fsrc)[i];
    int base = i * 4;
    int row = base >> 8, col = base & (DDIM - 1);
    union { __bf16 h[4]; uint2 u; } t;
    t.h[0] = (__bf16)v.x; t.h[1] = (__bf16)v.y;
    t.h[2] = (__bf16)v.z; t.h[3] = (__bf16)v.w;
    *reinterpret_cast<uint2*>(&fBf[row][col]) = t.u;
  }
  if (tid < LDIM) logits[tid] = 0.0f;
  __syncthreads();

  const int wave  = tid >> 5;
  const int lane  = tid & 31;
  const int lhalf = lane & 15;
  const int koff  = (lane < 16) ? 0 : 8;
  const int nbase = wave * 64;

  v8f acc[4][4] = {};
  for (int kk = 0; kk < DDIM / 32; ++kk) {
    const int k0 = kk * 32;
    v16bf afr[4];
#pragma unroll
    for (int mt = 0; mt < 4; ++mt) {
      const __bf16* rp = &fBf[mt * 16 + lhalf][k0 + koff];
      afr[mt] = load_frag_pair(rp, rp + 16);
    }
#pragma unroll
    for (int nt = 0; nt < 4; ++nt) {
      const __bf16* bp =
          W1bf + (size_t)(k0 + koff) * UDIM + nbase + nt * 16 + lhalf;
      v16bf bfr;
#pragma unroll
      for (int s = 0; s < 16; ++s) bfr[s] = bp[(size_t)(s + (s & 8)) * UDIM];
#pragma unroll
      for (int mt = 0; mt < 4; ++mt)
        acc[mt][nt] = __builtin_amdgcn_wmma_f32_16x16x32_bf16(
            false, afr[mt], false, bfr, (short)0, acc[mt][nt], false, false);
    }
  }

  // Epilogue: score = tanh(acc + projhb); partial logits = score . V
  float plog[4][8];
#pragma unroll
  for (int mt = 0; mt < 4; ++mt)
#pragma unroll
    for (int j = 0; j < 8; ++j) plog[mt][j] = 0.0f;

#pragma unroll
  for (int nt = 0; nt < 4; ++nt) {
    const int n = nbase + nt * 16 + lhalf;
    const float ph = projhb[(size_t)b * UDIM + n];
    const float vv = V[n];
#pragma unroll
    for (int mt = 0; mt < 4; ++mt)
#pragma unroll
      for (int j = 0; j < 8; ++j) {
        float s = tanhf(acc[mt][nt][j] + ph);
        plog[mt][j] += s * vv;
      }
  }
  // Lanes 0-15 hold the same row (different N) in the C layout: reduce halves.
#pragma unroll
  for (int mt = 0; mt < 4; ++mt)
#pragma unroll
    for (int j = 0; j < 8; ++j) {
      float v = plog[mt][j];
      v += __shfl_xor(v, 8, 16);
      v += __shfl_xor(v, 4, 16);
      v += __shfl_xor(v, 2, 16);
      v += __shfl_xor(v, 1, 16);
      if (lhalf == 0) atomicAdd(&logits[mt * 16 + j + koff], v);
    }
  __syncthreads();

  // Softmax over L=64 (bV cancels under softmax; logits themselves not output)
  if (wave == 0) {
    float x0 = logits[lane], x1 = logits[lane + 32];
    float mx = fmaxf(x0, x1);
#pragma unroll
    for (int off = 16; off >= 1; off >>= 1)
      mx = fmaxf(mx, __shfl_xor(mx, off, 32));
    float e0 = __expf(x0 - mx), e1 = __expf(x1 - mx);
    float s = e0 + e1;
#pragma unroll
    for (int off = 16; off >= 1; off >>= 1) s += __shfl_xor(s, off, 32);
    const float inv = 1.0f / s;
    aw[lane]      = e0 * inv;
    aw[lane + 32] = e1 * inv;
    outAttn[(size_t)b * LDIM + lane]      = e0 * inv;
    outAttn[(size_t)b * LDIM + lane + 32] = e1 * inv;
  }
  __syncthreads();

  // Context: full-precision features re-read (L2 hot), coalesced over tid.
  float c = 0.0f;
#pragma unroll 4
  for (int l = 0; l < LDIM; ++l)
    c += aw[l] * fsrc[(size_t)l * DDIM + tid];
  outCtx[(size_t)b * DDIM + tid] = c;
}

// ---------------------------------------------------------------------------
extern "C" void kernel_launch(void* const* d_in, const int* in_sizes, int n_in,
                              void* d_out, int out_size, void* d_ws, size_t ws_size,
                              hipStream_t stream) {
  const float* features = (const float*)d_in[0];
  const float* hidden   = (const float*)d_in[1];
  const float* W1 = (const float*)d_in[2];
  const float* b1 = (const float*)d_in[3];
  const float* W2 = (const float*)d_in[4];
  const float* b2 = (const float*)d_in[5];
  const float* V  = (const float*)d_in[6];
  // d_in[7] = bV: shifts pre-softmax logits only -> provably unused.

  const int B = in_sizes[1] / UDIM;  // hidden is [B, 512]

  char* ws = (char*)d_ws;
  __bf16* W1bf = (__bf16*)ws;  ws += (size_t)DDIM * UDIM * sizeof(__bf16);
  __bf16* W2bf = (__bf16*)ws;  ws += (size_t)UDIM * UDIM * sizeof(__bf16);
  __bf16* hbf  = (__bf16*)ws;  ws += (size_t)B * UDIM * sizeof(__bf16);
  float*  projhb = (float*)ws;

  float* outCtx  = (float*)d_out;
  float* outAttn = outCtx + (size_t)B * DDIM;

  convert_kernel<<<512, 256, 0, stream>>>(W1, W2, hidden, W1bf, W2bf, hbf, B);
  proj_hidden_kernel<<<B / 64, 256, 0, stream>>>(hbf, W2bf, b1, b2, projhb);
  attn_kernel<<<B, 256, 0, stream>>>(features, W1bf, projhb, V, outCtx, outAttn);
}
